// GatedCRFLoss_858993459294
// MI455X (gfx1250) — matrix operand order repaired and checked by reading, not compiled, not run on hardware
//
#include <hip/hip_runtime.h>

typedef __attribute__((ext_vector_type(16))) _Float16 v16h;
typedef __attribute__((ext_vector_type(8)))  float    v8f;

#define RR 5
#define HH 128
#define WW 128
#define CX 3
#define CY 21
#define NN 4
#define PLANE (HH * WW)                   // one channel plane, floats
#define TILES_X (WW / 16)                 // 8 tiles per row
#define NTILES  (NN * HH * TILES_X)       // 4096 wave-tiles
#define WAVES_PER_BLOCK 4

union AF { v16h v; _Float16 h[16]; };
union CF { v8f  v; float    f[8];  };

__device__ __forceinline__ int clampi(int v, int lo, int hi_) {
  return v < lo ? lo : (v > hi_ ? hi_ : v);
}

__global__ __launch_bounds__(128) void crf_main(const float* __restrict__ x,
                                                const float* __restrict__ y,
                                                float* __restrict__ partial) {
  const int lane = threadIdx.x & 31;
  const int tile = blockIdx.x * WAVES_PER_BLOCK + (threadIdx.x >> 5);
  const int n    = tile / (HH * TILES_X);
  const int rem  = tile % (HH * TILES_X);
  const int h    = rem / TILES_X;
  const int w0   = (rem % TILES_X) * 16;

  const int p  = lane & 15;    // pixel / column index within tile
  const int hi = lane >> 4;    // lane-half selector for WMMA K packing

  const float* yb = y + (size_t)n * CY * PLANE;
  const float* xb = x + (size_t)n * CX * PLANE;

  // ---- A fragment: 16 center pixels (row h, cols w0..w0+15) x 32 channels (f16).
  // ISA A layout: lanes 0-15 hold K {0..7,16..23}, lanes 16-31 hold K {8..15,24..31}.
  // Branchless: invalid channels (c>=21) read plane 0 at the same pixel, then cndmask to 0.
  AF a;
  const float* baseA = yb + ((size_t)(hi * 8) * HH + h) * WW + (w0 + p);
  #pragma unroll
  for (int e = 0; e < 16; ++e) {
    const int cofs = (e >= 8) ? (e + 8) : e;     // channel offset from hi*8
    const bool vld = (hi * 8 + cofs) < CY;
    const size_t off = vld ? (size_t)cofs * PLANE : 0;
    float v = baseA[off];
    a.h[e] = (_Float16)(vld ? v : 0.0f);
  }

  // ---- Per-lane center-x preload for the 8 M-rows this lane owns in the C tile.
  float xc0[8], xc1[8], xc2[8], oobA[8], oobB[8];
  #pragma unroll
  for (int r = 0; r < 8; ++r) {
    const int wc = w0 + 8 * hi + r;
    float c0 = xb[(size_t)h * WW + wc];
    float c1 = xb[(size_t)PLANE + (size_t)h * WW + wc];
    float c2 = xb[(size_t)2 * PLANE + (size_t)h * WW + wc];
    xc0[r] = c0; xc1[r] = c1; xc2[r] = c2;
    oobA[r] = (float)(wc * wc + h * h) * (1.0f / 36.0f);  // zero-pad: diff = -center
    oobB[r] = 100.0f * (c0 * c0 + c1 * c1 + c2 * c2);
  }

  float acc = 0.0f;

  for (int dy = -RR; dy <= RR; ++dy) {
    const int  hp    = h + dy;
    const int  hpc   = clampi(hp, 0, HH - 1);
    const bool rowOK = (hp >= 0) & (hp < HH);
    const float dy2  = (float)(dy * dy);

    #pragma unroll
    for (int t = 0; t < 2; ++t) {
      const int  qa    = w0 - 8 + 16 * t + p;   // neighbor column this lane owns
      const int  qac   = clampi(qa, 0, WW - 1);
      const bool colOK = rowOK & (qa >= 0) & (qa < WW);
      const float fm   = colOK ? 1.0f : 0.0f;   // single cndmask -> arithmetic blend mask

      // ---- B fragment: 32 channels x 16 neighbor pixels (row hp), branchless.
      // B layout (matches A): lanes 0-15 hold K {0..7,16..23}, lanes 16-31 hold K {8..15,24..31}.
      AF b;
      const float* baseB = yb + ((size_t)(hi * 8) * HH + hpc) * WW + qac;
      #pragma unroll
      for (int e = 0; e < 16; ++e) {
        const int cofs = (e >= 8) ? (e + 8) : e;
        const bool vld = (hi * 8 + cofs) < CY;
        const size_t off = vld ? (size_t)cofs * PLANE : 0;
        float v = baseB[off];
        b.h[e] = (_Float16)((vld & colOK) ? v : 0.0f);
      }

      // Neighbor x features at clamped (hpc, qac); blended out when !colOK.
      const float* xnb = xb + (size_t)hpc * WW + qac;
      const float xn0 = xnb[0];
      const float xn1 = xnb[PLANE];
      const float xn2 = xnb[2 * PLANE];

      // ---- Gram tile: G[m][q] = sum_c y_center[m,c] * y_neighbor[q,c] (f32 acc).
      CF g;
      #pragma unroll
      for (int i = 0; i < 8; ++i) g.f[i] = 0.0f;
      g.v = __builtin_amdgcn_wmma_f32_16x16x32_f16(
                false, a.v, false, b.v, (short)0, g.v, false, false);

      // ---- Consume band |dx|<=5; pure arithmetic dataflow, no selects of results.
      #pragma unroll
      for (int r = 0; r < 8; ++r) {
        const int m  = 8 * hi + r;
        const int dx = p - 8 - m + 16 * t;      // qa - (w0 + m)
        const float d0 = xn0 - xc0[r], d1 = xn1 - xc1[r], d2 = xn2 - xc2[r];
        const float Ain = ((float)(dx * dx) + dy2) * (1.0f / 36.0f);
        const float Bin = 100.0f * (d0 * d0 + d1 * d1 + d2 * d2);
        // blend in-bounds vs zero-pad forms with fma (cannot be turned into a branch)
        const float Aterm = fmaf(fm, Ain - oobA[r], oobA[r]);
        const float Bterm = fmaf(fm, Bin - oobB[r], oobB[r]);
        const float e1 = __expf(-0.5f * (Aterm + Bterm));
        const float e2 = __expf(-0.5f * Aterm);
        const float K  = fmaf(0.9f, e1, 0.1f * e2);
        // window + self-affinity mask as a float factor (cndmask of constants -> v_mul)
        const bool live = (dx >= -RR) & (dx <= RR) & ((dy != 0) | (dx != 0));
        const float lf  = live ? 1.0f : 0.0f;
        acc = fmaf(K * lf, 1.0f - g.f[r], acc); // Sum K - Sum K*dot
      }
    }
  }

  // Wave32 reduction, one partial per wave-tile (fully deterministic).
  #pragma unroll
  for (int off = 16; off > 0; off >>= 1)
    acc += __shfl_xor(acc, off, 32);

  if (lane == 0)
    partial[tile] = acc * (1.0f / (float)(NN * HH * WW));
}

__global__ __launch_bounds__(256) void crf_reduce(const float* __restrict__ partial,
                                                  float* __restrict__ out, int n) {
  __shared__ float s[256];
  float v = 0.0f;
  for (int i = threadIdx.x; i < n; i += 256) v += partial[i];  // fixed order
  s[threadIdx.x] = v;
  __syncthreads();
  for (int off = 128; off > 0; off >>= 1) {
    if ((int)threadIdx.x < off) s[threadIdx.x] += s[threadIdx.x + off];
    __syncthreads();
  }
  if (threadIdx.x == 0) out[0] = s[0];
}

extern "C" void kernel_launch(void* const* d_in, const int* in_sizes, int n_in,
                              void* d_out, int out_size, void* d_ws, size_t ws_size,
                              hipStream_t stream) {
  const float* x = (const float*)d_in[0];   // (4,3,128,128)
  const float* y = (const float*)d_in[1];   // (4,21,128,128)
  float* out     = (float*)d_out;           // scalar f32
  float* partial = (float*)d_ws;            // 4096 floats = 16 KB scratch

  dim3 block(32 * WAVES_PER_BLOCK);
  dim3 grid(NTILES / WAVES_PER_BLOCK);
  crf_main<<<grid, block, 0, stream>>>(x, y, partial);
  crf_reduce<<<1, 256, 0, stream>>>(partial, out, NTILES);
}
